// LQLinear_2061584302604
// MI455X (gfx1250) — matrix operand order repaired and checked by reading, not compiled, not run on hardware
//
#include <hip/hip_runtime.h>
#include <hip/hip_bf16.h>

#define TOKENS 8192
#define INF    4096   // K
#define OUTF   4096   // N

typedef __attribute__((ext_vector_type(16))) _Float16 v16h;
typedef __attribute__((ext_vector_type(8)))  _Float16 v8h;
typedef __attribute__((ext_vector_type(4)))  _Float16 v4h;
typedef __attribute__((ext_vector_type(8)))  float    v8f;
typedef __attribute__((ext_vector_type(4)))  float    v4f;
typedef __attribute__((ext_vector_type(4)))  unsigned int u32x4;
typedef __attribute__((ext_vector_type(8)))  int      i32x8;
typedef __attribute__((ext_vector_type(4)))  int      i32x4;

// ---------------- x: f32 -> f16 ----------------
__global__ __launch_bounds__(256)
void cvt_x_f16(const float* __restrict__ x, _Float16* __restrict__ xh, int n4) {
  int i = blockIdx.x * blockDim.x + threadIdx.x;
  if (i >= n4) return;
  v4f v = ((const v4f*)x)[i];
  v4h h;
  h[0] = (_Float16)v[0]; h[1] = (_Float16)v[1];
  h[2] = (_Float16)v[2]; h[3] = (_Float16)v[3];
  ((v4h*)xh)[i] = h;
}

// ---------------- W: quantize to 4 levels, emit f16 ----------------
__device__ __forceinline__ void cswap(float& a, float& b) {
  float lo = fminf(a, b), hi = fmaxf(a, b);
  a = lo; b = hi;
}

__global__ __launch_bounds__(256)
void quant_w(const float* __restrict__ w, const float* __restrict__ basis,
             _Float16* __restrict__ wh, int n4) {
  int i = blockIdx.x * blockDim.x + threadIdx.x;
  if (i >= n4) return;
  const float b0 = basis[0], b1 = basis[1];
  float l0 = -b0 - b1, l1 = -b0 + b1, l2 = b0 - b1, l3 = b0 + b1;
  cswap(l0, l1); cswap(l2, l3); cswap(l0, l2); cswap(l1, l3); cswap(l1, l2);
  const float t0 = 0.5f * (l0 + l1);
  const float t1 = 0.5f * (l1 + l2);
  const float t2 = 0.5f * (l2 + l3);
  v4f v = ((const v4f*)w)[i];
  v4h q;
#pragma unroll
  for (int j = 0; j < 4; ++j) {
    float wv = v[j];
    int idx = (wv > t0) + (wv > t1) + (wv > t2);  // searchsorted(side=left)
    float lv = (idx == 0) ? l0 : (idx == 1) ? l1 : (idx == 2) ? l2 : l3;
    q[j] = (_Float16)lv;
  }
  ((v4h*)wh)[i] = q;
}

// ---------------- TDM: DMA one 128-row x 64-half tile into LDS ----------------
// D# per CDNA5 ISA. 2D tile, data_size=2B,
// pad_enable: +16B after every 128B row -> LDS row stride 72 halves (144B).
__device__ __forceinline__ void tdm_load_tile(unsigned lds_off, const _Float16* gptr) {
  unsigned long long ga = (unsigned long long)(uintptr_t)gptr;
  u32x4 g0;
  g0[0] = 1u;                                             // count=1, user desc
  g0[1] = lds_off;                                        // lds_addr (bytes)
  g0[2] = (unsigned)ga;                                   // global_addr[31:0]
  g0[3] = (unsigned)((ga >> 32) & 0x1FFFFFFu) | (2u << 30); // addr[56:32] | type=2
  i32x8 g1;
  g1[0] = (int)((1u << 16)    // data_size = 2 bytes
              | (1u << 20)    // pad_enable
              | (4u << 22)    // pad_interval: 32 DWORDs (128B) between pads
              | (3u << 25));  // pad_amount: 4 DWORDs (16B)
  g1[1] = (int)(4096u << 16); // tensor_dim0 = 4096 (K, in halves), low bits
  g1[2] = (int)(128u << 16);  // tensor_dim0 hi=0 | tensor_dim1 = 128 rows
  g1[3] = (int)(64u << 16);   // tensor_dim1 hi=0 | tile_dim0 = 64 halves
  g1[4] = 128;                // tile_dim1 = 128 rows | tile_dim2 = 0
  g1[5] = 4096;               // tensor_dim0_stride = 4096 halves (row pitch)
  g1[6] = 0;
  g1[7] = 0;
  i32x4 z4 = {0, 0, 0, 0};    // groups 2/3 unused for 2D tile
#if __clang_major__ >= 23
  i32x8 z8 = {};
  __builtin_amdgcn_tensor_load_to_lds(g0, g1, z4, z4, z8, 0);
#else
  __builtin_amdgcn_tensor_load_to_lds(g0, g1, z4, z4, 0);
#endif
}

// ---------------- GEMM: out[M,N] = Xh[M,K] * Wh[N,K]^T + bias ----------------
// Block tile 128x128, 256 threads = 8 waves (wave32), K-step 64.
// Double-buffered LDS tiles filled by TDM (wave 0 issues, TENSORcnt tracked).
__global__ __launch_bounds__(256)
void lq_gemm(const _Float16* __restrict__ A,   // [TOKENS x INF] f16 row-major
             const _Float16* __restrict__ B,   // [OUTF x INF] f16 row-major
             const float* __restrict__ bias,   // [OUTF]
             float* __restrict__ C) {          // [TOKENS x OUTF] f32
  constexpr int K     = INF;
  constexpr int N     = OUTF;
  constexpr int KSTEP = 64;
  constexpr int LDT   = 72;   // 64 data halves + 8 pad halves (TDM pad) per row

  __shared__ _Float16 lA[2][128 * LDT];
  __shared__ _Float16 lB[2][128 * LDT];

  const int t     = threadIdx.x;
  const int lane  = t & 31;
  const int w     = t >> 5;
  const int waveM = w >> 2;        // 0..1 -> 64 rows each
  const int waveN = w & 3;         // 0..3 -> 32 cols each
  const int lhalf = lane >> 4;
  const int lrow  = lane & 15;

  const size_t row0 = (size_t)blockIdx.y * 128;  // M base
  const size_t n0   = (size_t)blockIdx.x * 128;  // N base

  const unsigned ldsA0 = (unsigned)(uintptr_t)&lA[0][0];
  const unsigned ldsA1 = (unsigned)(uintptr_t)&lA[1][0];
  const unsigned ldsB0 = (unsigned)(uintptr_t)&lB[0][0];
  const unsigned ldsB1 = (unsigned)(uintptr_t)&lB[1][0];

  v8f acc[4][2];
#pragma unroll
  for (int i = 0; i < 4; ++i)
#pragma unroll
    for (int j = 0; j < 2; ++j) acc[i][j] = v8f{};

  // prologue: fill buffer 0
  if (w == 0) {
    tdm_load_tile(ldsA0, A + row0 * K);
    tdm_load_tile(ldsB0, B + n0 * K);
    __builtin_amdgcn_s_wait_tensorcnt(0);
  }
  __syncthreads();

  for (int kk = 0; kk < K; kk += KSTEP) {
    const int cur = (kk >> 6) & 1;
    // kick off DMA for the next buffer; overlaps with WMMA below
    if (w == 0 && (kk + KSTEP) < K) {
      tdm_load_tile(cur ? ldsA0 : ldsA1, A + row0 * K + kk + KSTEP);
      tdm_load_tile(cur ? ldsB0 : ldsB1, B + n0 * K + kk + KSTEP);
    }
    const _Float16* sA = &lA[cur][0];
    const _Float16* sB = &lB[cur][0];

#pragma unroll
    for (int ks = 0; ks < 2; ++ks) {     // two 32-wide WMMA K-slices
      v16h af[4];
#pragma unroll
      for (int tm = 0; tm < 4; ++tm) {
        const int r = waveM * 64 + tm * 16 + lrow;
        const int base = r * LDT + ks * 32;
        v8h x0 = *(const v8h*)&sA[base + lhalf * 8];
        v8h x1 = *(const v8h*)&sA[base + 16 + lhalf * 8];
        af[tm] = __builtin_shufflevector(x0, x1, 0, 1, 2, 3, 4, 5, 6, 7,
                                                 8, 9, 10, 11, 12, 13, 14, 15);
      }
      v16h bf[2];
#pragma unroll
      for (int tn = 0; tn < 2; ++tn) {
        const int r = waveN * 32 + tn * 16 + lrow;
        const int base = r * LDT + ks * 32;
        v8h y0 = *(const v8h*)&sB[base + lhalf * 8];
        v8h y1 = *(const v8h*)&sB[base + 16 + lhalf * 8];
        bf[tn] = __builtin_shufflevector(y0, y1, 0, 1, 2, 3, 4, 5, 6, 7,
                                                 8, 9, 10, 11, 12, 13, 14, 15);
      }
#pragma unroll
      for (int tm = 0; tm < 4; ++tm)
#pragma unroll
        for (int tn = 0; tn < 2; ++tn)
          acc[tm][tn] = __builtin_amdgcn_wmma_f32_16x16x32_f16(
              false, af[tm], false, bf[tn], (short)0, acc[tm][tn], false, false);
    }

    // next buffer must be resident before anyone reads it next iteration;
    // one barrier also protects cur^1 from being overwritten while still read
    if (w == 0) __builtin_amdgcn_s_wait_tensorcnt(0);
    __syncthreads();
  }

  // epilogue: C layout — VGPR r holds M = r + 8*lhalf, N = lrow
#pragma unroll
  for (int tm = 0; tm < 4; ++tm) {
    const size_t m0 = row0 + waveM * 64 + tm * 16 + lhalf * 8;
#pragma unroll
    for (int tn = 0; tn < 2; ++tn) {
      const size_t n = n0 + waveN * 32 + tn * 16 + lrow;
      const float bv = bias[n];
      v8f c = acc[tm][tn];
#pragma unroll
      for (int r = 0; r < 8; ++r) C[(m0 + r) * N + n] = c[r] + bv;
    }
  }
}

extern "C" void kernel_launch(void* const* d_in, const int* in_sizes, int n_in,
                              void* d_out, int out_size, void* d_ws, size_t ws_size,
                              hipStream_t stream) {
  const float* x     = (const float*)d_in[0];  // [8192 x 4096]
  const float* wgt   = (const float*)d_in[1];  // [4096 x 4096]
  const float* bias  = (const float*)d_in[2];  // [4096]
  const float* basis = (const float*)d_in[3];  // [2]
  float* out = (float*)d_out;

  _Float16* xh = (_Float16*)d_ws;                        // 64 MB
  _Float16* wh = xh + (size_t)TOKENS * INF;              // +32 MB

  const int nx4 = TOKENS * INF / 4;
  cvt_x_f16<<<nx4 / 256, 256, 0, stream>>>(x, xh, nx4);

  const int nw4 = OUTF * INF / 4;
  quant_w<<<nw4 / 256, 256, 0, stream>>>(wgt, basis, wh, nw4);

  dim3 grid(OUTF / 128, TOKENS / 128);
  lq_gemm<<<grid, 256, 0, stream>>>(xh, wh, bias, out);
}
